// GiG_19481971655359
// MI455X (gfx1250) — matrix operand order repaired and checked by reading, not compiled, not run on hardware
//
#include <hip/hip_runtime.h>

#define NN 65536
#define EE 524288
#define BB 2048
#define FIN 128
#define HH 256
#define PH 128

typedef __bf16 bf16;
typedef __attribute__((ext_vector_type(16))) __bf16 bf16x16;
typedef __attribute__((ext_vector_type(8)))  float  f32x8;
typedef __attribute__((ext_vector_type(4)))  unsigned int u32x4;

union FragB { u32x4 u[2]; bf16x16 v; };

__device__ __forceinline__ float lrelu(float v) { return v > 0.0f ? v : 0.01f * v; }

// ---------------- f32 -> bf16 (row-major copy, for A operands) ----------------
__global__ void k_cvt(const float* __restrict__ in, bf16* __restrict__ out, int n) {
  int i = blockIdx.x * blockDim.x + threadIdx.x;
  if (i < n) out[i] = (bf16)in[i];
}

// ---------------- f32 [K x Nc] -> bf16 [Nc x K] (transpose-pack for B operands) ----------------
__global__ void k_cvt_t(const float* __restrict__ in, bf16* __restrict__ out, int K, int Nc) {
  int i = blockIdx.x * blockDim.x + threadIdx.x;
  if (i >= K * Nc) return;
  int k = i / Nc, n = i - k * Nc;
  out[(size_t)n * K + k] = (bf16)in[i];
}

// ---------------- WMMA GEMM ----------------
// C[M x Nc](f32) = act( A1 @ Wt1^T  (+ A2 @ Wt2^T)  + bias )
// A: row-major [M x K] bf16.  Wt: TRANSPOSED row-major [Nc x K] bf16 (contiguous K).
// One wave computes a 16x64 output strip: 4 N-sub-tiles share one A fragment per K-step.
// FULL=true: Nc % 64 == 0 and M % 16 == 0 -> no bounds checks, no divergence at all.
template <bool FULL>
__global__ void k_gemm(const bf16* __restrict__ A1, const bf16* __restrict__ Wt1, int K1,
                       const bf16* __restrict__ A2, const bf16* __restrict__ Wt2, int K2,
                       const float* __restrict__ bias, float* __restrict__ C,
                       int M, int Nc, int tiles_ng, int act)
{
  const int wave = threadIdx.x >> 5;
  const int lane = threadIdx.x & 31;
  const int tile = blockIdx.x * (blockDim.x >> 5) + wave;
  const int tiles_m = M >> 4;
  if (tile >= tiles_m * tiles_ng) return;         // wave-uniform
  const int mt  = tile / tiles_ng;
  const int ntg = tile - mt * tiles_ng;           // group of 4 n-tiles
  const int half = lane >> 4;                     // K sub-block select
  const int l15  = lane & 15;                     // m (for A) / n (for B,C)

  int  colg[4];
  bool colOk[4];
  #pragma unroll
  for (int s = 0; s < 4; ++s) {
    colg[s]  = (ntg * 4 + s) * 16 + l15;
    colOk[s] = FULL ? true : (colg[s] < Nc);
  }

  f32x8 acc[4] = {{}, {}, {}, {}};

  // ---- product 1 ----
  {
    const bf16* arow = A1 + (size_t)(mt * 16 + l15) * K1;
    for (int k0 = 0; k0 < K1; k0 += 32) {
      FragB a;
      // A 16-bit layout: elems 0..7 -> K = k0+half*8+e ; elems 8..15 -> K = k0+16+half*8+(e-8)
      a.u[0] = *(const u32x4*)(arow + k0 + half * 8);
      a.u[1] = *(const u32x4*)(arow + k0 + 16 + half * 8);
      #pragma unroll
      for (int s = 0; s < 4; ++s) {
        FragB b;
        if (FULL || colOk[s]) {                   // B: same K layout as A (half*8, 16+half*8)
          const bf16* wr = Wt1 + (size_t)colg[s] * K1 + k0 + half * 8;
          b.u[0] = *(const u32x4*)(wr);
          b.u[1] = *(const u32x4*)(wr + 16);
        } else {
          b.u[0] = (u32x4)0u;
          b.u[1] = (u32x4)0u;
        }
        acc[s] = __builtin_amdgcn_wmma_f32_16x16x32_bf16(false, a.v, false, b.v,
                                                         (short)0, acc[s], false, false);
      }
    }
  }
  // ---- optional product 2 (GraphConv lin_root) ----
  if (A2 != nullptr) {
    const bf16* arow = A2 + (size_t)(mt * 16 + l15) * K2;
    for (int k0 = 0; k0 < K2; k0 += 32) {
      FragB a;
      a.u[0] = *(const u32x4*)(arow + k0 + half * 8);
      a.u[1] = *(const u32x4*)(arow + k0 + 16 + half * 8);
      #pragma unroll
      for (int s = 0; s < 4; ++s) {
        FragB b;
        if (FULL || colOk[s]) {
          const bf16* wr = Wt2 + (size_t)colg[s] * K2 + k0 + half * 8;
          b.u[0] = *(const u32x4*)(wr);
          b.u[1] = *(const u32x4*)(wr + 16);
        } else {
          b.u[0] = (u32x4)0u;
          b.u[1] = (u32x4)0u;
        }
        acc[s] = __builtin_amdgcn_wmma_f32_16x16x32_bf16(false, a.v, false, b.v,
                                                         (short)0, acc[s], false, false);
      }
    }
  }

  #pragma unroll
  for (int s = 0; s < 4; ++s) {
    const float bv = (bias != nullptr && (FULL || colOk[s])) ? bias[colg[s]] : 0.0f;
    #pragma unroll
    for (int r = 0; r < 8; ++r) {                 // C: row m = r + 8*half, col = l15
      const int row = mt * 16 + half * 8 + r;
      float v = acc[s][r] + bv;
      if (act) v = lrelu(v);
      if (FULL || (colOk[s] && row < M)) C[(size_t)row * Nc + colg[s]] = v;
    }
  }
}

// ---------------- edge scatter-add: agg[dst] += h[src] ----------------
__global__ void k_scatter(const float* __restrict__ h, const int* __restrict__ src,
                          const int* __restrict__ dst, float* __restrict__ agg,
                          long long total)
{
  long long i = (long long)blockIdx.x * blockDim.x + threadIdx.x;
  if (i >= total) return;
  int e = (int)(i >> 8);          // / HH (HH == 256)
  int c = (int)(i & 255);         // % HH
  atomicAdd(&agg[(size_t)dst[e] * HH + c], h[(size_t)src[e] * HH + c]);
}

// ---------------- row LayerNorm(256) + LeakyReLU (+ residual) ----------------
__global__ void k_ln(const float* __restrict__ T, const float* __restrict__ g,
                     const float* __restrict__ b, const float* __restrict__ res,
                     float* __restrict__ out, int rows)
{
  int row  = blockIdx.x * (blockDim.x >> 5) + (threadIdx.x >> 5);
  int lane = threadIdx.x & 31;
  if (row >= rows) return;
  const float* t = T + (size_t)row * HH;
  float v[8];
  float s = 0.0f;
  #pragma unroll
  for (int j = 0; j < 8; ++j) { v[j] = t[j * 32 + lane]; s += v[j]; }
  #pragma unroll
  for (int off = 16; off > 0; off >>= 1) s += __shfl_xor(s, off, 32);
  float mean = s * (1.0f / 256.0f);
  float s2 = 0.0f;
  #pragma unroll
  for (int j = 0; j < 8; ++j) { float d = v[j] - mean; s2 += d * d; }
  #pragma unroll
  for (int off = 16; off > 0; off >>= 1) s2 += __shfl_xor(s2, off, 32);
  float rstd = rsqrtf(s2 * (1.0f / 256.0f) + 1e-5f);
  #pragma unroll
  for (int j = 0; j < 8; ++j) {
    int c = j * 32 + lane;
    float y = (v[j] - mean) * rstd * g[c] + b[c];
    y = lrelu(y);
    if (res != nullptr) y += res[(size_t)row * HH + c];
    out[(size_t)row * HH + c] = y;
  }
}

// ---------------- mean pool: 32 sorted nodes per graph ----------------
__global__ void k_pool(const float* __restrict__ h2, float* __restrict__ feat) {
  int i = blockIdx.x * blockDim.x + threadIdx.x;   // i = b*HH + c
  if (i >= BB * HH) return;
  int bg = i >> 8, c = i & 255;
  const float* p = h2 + (size_t)bg * 32 * HH + c;
  float s = 0.0f;
  #pragma unroll
  for (int k = 0; k < 32; ++k) s += p[k * HH];
  feat[i] = s * (1.0f / 32.0f);
}

// ---------------- per-row squared norm of z2 [BB x PH] ----------------
__global__ void k_sq(const float* __restrict__ z, float* __restrict__ sq) {
  int i = blockIdx.x * blockDim.x + threadIdx.x;
  if (i >= BB) return;
  const float* p = z + (size_t)i * PH;
  float s = 0.0f;
  for (int k = 0; k < PH; ++k) s += p[k] * p[k];
  sq[i] = s;
}

// ---------------- adjacency: sigmoid(temp * -sqrt(d2+eps)*mask + theta + eye) ----------------
__global__ void k_adj(const float* __restrict__ gram, const float* __restrict__ sq,
                      const float* __restrict__ temp_p, const float* __restrict__ theta_p,
                      bf16* __restrict__ adjB)
{
  int i = blockIdx.x * blockDim.x + threadIdx.x;
  if (i >= BB * BB) return;
  int r = i >> 11, c = i & 2047;                   // BB == 2048
  float d2 = sq[r] + sq[c] - 2.0f * gram[i];
  d2 = fmaxf(d2, 0.0f);
  if (r == c) d2 = 0.0f;                           // exact-zero diagonal
  float mask = (d2 != 0.0f) ? 1.0f : 0.0f;
  float dist = -sqrtf(d2 + 1.1920928955078125e-07f) * mask;
  float eye  = (r == c) ? 1.0f : 0.0f;
  float a = temp_p[0] * dist + theta_p[0] + eye;
  adjB[i] = (bf16)(1.0f / (1.0f + expf(-a)));
}

// ---------------- host orchestration ----------------
extern "C" void kernel_launch(void* const* d_in, const int* in_sizes, int n_in,
                              void* d_out, int out_size, void* d_ws, size_t ws_size,
                              hipStream_t stream)
{
  (void)in_sizes; (void)n_in; (void)out_size; (void)ws_size;
  const float* x        = (const float*)d_in[0];
  const int*   ei       = (const int*)d_in[1];
  const int*   srcp     = ei;
  const int*   dstp     = ei + EE;
  const float* proj_w1  = (const float*)d_in[3];
  const float* proj_b1  = (const float*)d_in[4];
  const float* proj_w2  = (const float*)d_in[5];
  const float* proj_b2  = (const float*)d_in[6];
  const float* c1_wrel  = (const float*)d_in[7];
  const float* c1_wroot = (const float*)d_in[8];
  const float* c1_b     = (const float*)d_in[9];
  const float* ln1_g    = (const float*)d_in[10];
  const float* ln1_b    = (const float*)d_in[11];
  const float* c2_wrel  = (const float*)d_in[12];
  const float* c2_wroot = (const float*)d_in[13];
  const float* c2_b     = (const float*)d_in[14];
  const float* ln2_g    = (const float*)d_in[15];
  const float* ln2_b    = (const float*)d_in[16];
  const float* lgl_w1   = (const float*)d_in[17];
  const float* lgl_b1   = (const float*)d_in[18];
  const float* lgl_w2   = (const float*)d_in[19];
  const float* lgl_b2   = (const float*)d_in[20];
  const float* temp     = (const float*)d_in[21];
  const float* theta    = (const float*)d_in[22];
  const float* g_wrel   = (const float*)d_in[23];
  const float* g_wroot  = (const float*)d_in[24];
  const float* g_b      = (const float*)d_in[25];
  const float* cls_w1   = (const float*)d_in[26];
  const float* cls_b1   = (const float*)d_in[27];
  const float* cls_w2   = (const float*)d_in[28];
  const float* cls_b2   = (const float*)d_in[29];

  // carve workspace
  char* p = (char*)d_ws;
  auto take = [&](size_t bytes) -> void* {
    void* r = (void*)p;
    p += (bytes + 255) & ~(size_t)255;
    return r;
  };
  float* bufA = (float*)take((size_t)NN * HH * 4);
  float* bufB = (float*)take((size_t)NN * HH * 4);
  float* bufC = (float*)take((size_t)NN * HH * 4);
  float* gram = (float*)take((size_t)BB * BB * 4);
  float* feat = (float*)take((size_t)BB * HH * 4);
  float* z1   = (float*)take((size_t)BB * HH * 4);
  float* z2   = (float*)take((size_t)BB * PH * 4);
  float* sqv  = (float*)take((size_t)BB * 4);
  float* aggP = (float*)take((size_t)BB * HH * 4);
  float* gbuf = (float*)take((size_t)BB * HH * 4);
  float* c1v  = (float*)take((size_t)BB * PH * 4);
  bf16* xB    = (bf16*)take((size_t)NN * FIN * 2);
  bf16* stg1  = (bf16*)take((size_t)NN * HH * 2);
  bf16* stg2  = (bf16*)take((size_t)NN * HH * 2);
  bf16* adjB  = (bf16*)take((size_t)BB * BB * 2);
  bf16* featB = (bf16*)take((size_t)BB * HH * 2);   // A-side layout [BB x HH]
  bf16* featT = (bf16*)take((size_t)BB * HH * 2);   // B-side layout [HH x BB]
  bf16* z1B   = (bf16*)take((size_t)BB * HH * 2);
  bf16* z2B   = (bf16*)take((size_t)BB * PH * 2);
  bf16* aggPB = (bf16*)take((size_t)BB * HH * 2);
  bf16* gB    = (bf16*)take((size_t)BB * HH * 2);
  bf16* c1B   = (bf16*)take((size_t)BB * PH * 2);
  // transposed-packed weights [Nc x K]
  bf16* w_p1  = (bf16*)take((size_t)FIN * HH * 2);
  bf16* w_p2  = (bf16*)take((size_t)HH * HH * 2);
  bf16* w_c1r = (bf16*)take((size_t)HH * HH * 2);
  bf16* w_c1o = (bf16*)take((size_t)HH * HH * 2);
  bf16* w_c2r = (bf16*)take((size_t)HH * HH * 2);
  bf16* w_c2o = (bf16*)take((size_t)HH * HH * 2);
  bf16* w_l1  = (bf16*)take((size_t)HH * HH * 2);
  bf16* w_l2  = (bf16*)take((size_t)HH * PH * 2);
  bf16* w_gr  = (bf16*)take((size_t)HH * HH * 2);
  bf16* w_go  = (bf16*)take((size_t)HH * HH * 2);
  bf16* w_cl1 = (bf16*)take((size_t)HH * PH * 2);
  bf16* w_cl2 = (bf16*)take((size_t)PH * 2);

  auto cvt = [&](const float* in, bf16* out, int n) {
    k_cvt<<<(n + 255) / 256, 256, 0, stream>>>(in, out, n);
  };
  auto cvt_t = [&](const float* in, bf16* out, int K, int Nc) {
    k_cvt_t<<<(K * Nc + 255) / 256, 256, 0, stream>>>(in, out, K, Nc);
  };
  auto gemm = [&](const bf16* A1, const bf16* Wt1, int K1,
                  const bf16* A2, const bf16* Wt2, int K2,
                  const float* bias, float* C, int M, int Nc, int act) {
    int tiles_ng = (Nc + 63) >> 6;                 // groups of 4 n-tiles
    int tiles = (M >> 4) * tiles_ng;
    int blocks = (tiles + 7) >> 3;
    if ((Nc & 63) == 0 && (M & 15) == 0) {
      k_gemm<true><<<blocks, 256, 0, stream>>>(A1, Wt1, K1, A2, Wt2, K2,
                                               bias, C, M, Nc, tiles_ng, act);
    } else {
      k_gemm<false><<<blocks, 256, 0, stream>>>(A1, Wt1, K1, A2, Wt2, K2,
                                                bias, C, M, Nc, tiles_ng, act);
    }
  };

  // weights -> transposed bf16 [Nc x K]
  cvt_t(proj_w1, w_p1, FIN, HH);   cvt_t(proj_w2, w_p2, HH, HH);
  cvt_t(c1_wrel, w_c1r, HH, HH);   cvt_t(c1_wroot, w_c1o, HH, HH);
  cvt_t(c2_wrel, w_c2r, HH, HH);   cvt_t(c2_wroot, w_c2o, HH, HH);
  cvt_t(lgl_w1, w_l1, HH, HH);     cvt_t(lgl_w2, w_l2, HH, PH);
  cvt_t(g_wrel, w_gr, HH, HH);     cvt_t(g_wroot, w_go, HH, HH);
  cvt_t(cls_w1, w_cl1, HH, PH);    cvt_t(cls_w2, w_cl2, PH, 1);

  const long long scat_total = (long long)EE * HH;
  const int scat_blocks = (int)((scat_total + 255) / 256);

  // projection MLP
  cvt(x, xB, NN * FIN);
  gemm(xB, w_p1, FIN, nullptr, nullptr, 0, proj_b1, bufA, NN, HH, 1);      // h = leaky(x@W1+b1)
  cvt(bufA, stg1, NN * HH);
  gemm(stg1, w_p2, HH, nullptr, nullptr, 0, proj_b2, bufB, NN, HH, 1);     // h = leaky(h@W2+b2)

  // GraphConv 1
  cvt(bufB, stg1, NN * HH);                                                // h (bf16)
  hipMemsetAsync(bufA, 0, (size_t)NN * HH * 4, stream);
  k_scatter<<<scat_blocks, 256, 0, stream>>>(bufB, srcp, dstp, bufA, scat_total);
  cvt(bufA, stg2, NN * HH);                                                // agg (bf16)
  gemm(stg2, w_c1r, HH, stg1, w_c1o, HH, c1_b, bufC, NN, HH, 0);
  k_ln<<<NN / 8, 256, 0, stream>>>(bufC, ln1_g, ln1_b, nullptr, bufA, NN); // h1

  // GraphConv 2 (+ residual)
  cvt(bufA, stg1, NN * HH);                                                // h1 (bf16)
  hipMemsetAsync(bufB, 0, (size_t)NN * HH * 4, stream);
  k_scatter<<<scat_blocks, 256, 0, stream>>>(bufA, srcp, dstp, bufB, scat_total);
  cvt(bufB, stg2, NN * HH);                                                // agg2 (bf16)
  gemm(stg2, w_c2r, HH, stg1, w_c2o, HH, c2_b, bufC, NN, HH, 0);
  k_ln<<<NN / 8, 256, 0, stream>>>(bufC, ln2_g, ln2_b, bufA, bufB, NN);    // h2 = LN.. + h1

  // pooling -> population features (A layout + B layout)
  k_pool<<<(BB * HH) / 256, 256, 0, stream>>>(bufB, feat);
  cvt(feat, featB, BB * HH);
  cvt_t(feat, featT, BB, HH);                      // feat as [K=BB x Nc=HH] -> [HH x BB]

  // LGL embedding
  gemm(featB, w_l1, HH, nullptr, nullptr, 0, lgl_b1, z1, BB, HH, 1);
  cvt(z1, z1B, BB * HH);
  gemm(z1B, w_l2, HH, nullptr, nullptr, 0, lgl_b2, z2, BB, PH, 1);
  k_sq<<<(BB + 255) / 256, 256, 0, stream>>>(z2, sqv);
  cvt(z2, z2B, BB * PH);
  gemm(z2B, z2B, PH, nullptr, nullptr, 0, nullptr, gram, BB, BB, 0);       // z2 @ z2^T
  k_adj<<<(BB * BB) / 256, 256, 0, stream>>>(gram, sqv, temp, theta, adjB);

  // population GNN (adj symmetric => adj.T @ feat == adj @ feat)
  gemm(adjB, featT, BB, nullptr, nullptr, 0, nullptr, aggP, BB, HH, 0);
  cvt(aggP, aggPB, BB * HH);
  gemm(aggPB, w_gr, HH, featB, w_go, HH, g_b, gbuf, BB, HH, 1);

  // classifier
  cvt(gbuf, gB, BB * HH);
  gemm(gB, w_cl1, HH, nullptr, nullptr, 0, cls_b1, c1v, BB, PH, 1);
  cvt(c1v, c1B, BB * PH);
  gemm(c1B, w_cl2, PH, nullptr, nullptr, 0, cls_b2, (float*)d_out, BB, 1, 0);
}
